// HConvolution2d_88424786690696
// MI455X (gfx1250) — compile-verified, matrix operand order checked by reading, not run on hardware
//
#include <hip/hip_runtime.h>
#include <cmath>

typedef __attribute__((ext_vector_type(8)))  float          v8f;
typedef __attribute__((ext_vector_type(16))) __bf16         v16bf;
typedef __attribute__((ext_vector_type(8)))  unsigned short v8us;
typedef __attribute__((ext_vector_type(16))) unsigned short v16us;

#define NB    16
#define CIN   256
#define COUT  256
#define HW    4096
#define KTOT  2304   // 9 taps * 256 channels (tap-major after reorder)
#define BSTR  48     // padded B row stride in halves (96 B, keeps 32B alignment)

__device__ __forceinline__ unsigned short f2bf(float f) {
  unsigned u = __float_as_uint(f);
  u += 0x7FFFu + ((u >> 16) & 1u);          // round-to-nearest-even
  return (unsigned short)(u >> 16);
}

// ---------------- prep 1: logmap0 + beta scale -> bf16 v, per-pixel ||v||^2 ----
__global__ __launch_bounds__(256) void prep_pixels(const float* __restrict__ x,
                                                   unsigned short* __restrict__ vbf,
                                                   float* __restrict__ s,
                                                   float ratio) {
  int p  = blockIdx.x * 256 + threadIdx.x;   // 65536 pixels
  int b  = p >> 12;
  int hw = p & 4095;
  int base = (b << 20) + hw;                 // b*CIN*HW + hw
  float ss = 0.f;
#pragma unroll 4
  for (int c = 0; c < CIN; ++c) {
    float v = x[base + (c << 12)];
    ss += v * v;
  }
  float n   = sqrtf(ss);
  float nm  = fmaxf(n, 1e-15f);
  float arg = fminf(nm, 1.0f - 1e-7f);
  float at  = 0.5f * logf((1.0f + arg) / (1.0f - arg));   // atanh
  float sc  = at / nm * ratio;               // logmap0 scale * BETA_N/BETA_NI
#pragma unroll 4
  for (int c = 0; c < CIN; ++c)
    vbf[base + (c << 12)] = f2bf(x[base + (c << 12)] * sc);
  s[p] = ss * sc * sc;
}

// ---------------- prep 2: weights -> bf16, tap-major, transposed [co][tap*256+c]
__global__ __launch_bounds__(256) void prep_weights(const float* __restrict__ z,
                                                    unsigned short* __restrict__ Wt) {
  int k  = blockIdx.x;                       // 0..2303, original row = c*9+tap
  int co = threadIdx.x;
  int c   = k / 9;
  int tap = k - c * 9;
  Wt[co * KTOT + tap * 256 + c] = f2bf(z[k * 256 + co]);
}

// ---------------- prep 3: column norms of z (two-stage, no atomics) -----------
__global__ __launch_bounds__(256) void znorm_part(const float* __restrict__ z,
                                                  float* __restrict__ part) {
  int j = blockIdx.x, co = threadIdx.x;      // 72 blocks x 32 rows
  float a = 0.f;
#pragma unroll 1
  for (int r = 0; r < 32; ++r) {
    float v = z[(j * 32 + r) * 256 + co];
    a += v * v;
  }
  part[j * 256 + co] = a;
}
__global__ __launch_bounds__(256) void znorm_fin(const float* __restrict__ part,
                                                 float* __restrict__ zn2) {
  int co = threadIdx.x;
  float a = 0.f;
#pragma unroll 1
  for (int j = 0; j < 72; ++j) a += part[j * 256 + co];
  zn2[co] = a;
}

// ---------------- fused hyperbolic conv: WMMA GEMM + epilogue -----------------
// K order inside each 32-group is permuted at staging: 8-blocks stored 0,2,1,3
// so a lane's ISA fragment halves {klo..klo+7, klo+16..klo+23} are contiguous.
__device__ __forceinline__ int permblk(int blk) {           // 0,1,2,3 -> 0,2,1,3
  return ((blk & 1) << 1) | (blk >> 1);
}

__global__ __launch_bounds__(256) void hconv_wmma(const unsigned short* __restrict__ vbf,
                                                  const float* __restrict__ s,
                                                  const unsigned short* __restrict__ Wt,
                                                  const float* __restrict__ zn2,
                                                  const float* __restrict__ bias,
                                                  float* __restrict__ out) {
  // LDS: 3x66x256 bf16 halo (A, reused as fp32 w-buffer), 256x48 bf16 B chunk
  __shared__ __align__(32) unsigned short s_v[3 * 66 * 256];   // 101376 B
  __shared__ __align__(32) unsigned short s_b[256 * BSTR];     //  24576 B
  __shared__ float s_z2[256], s_chzn[256], s_sh[256];
  __shared__ float s_t[64], s_lam[64], s_dinv[64];
  __shared__ float s_ps[256];

  const int tid  = threadIdx.x;
  const int bimg = blockIdx.x >> 6;          // batch
  const int h    = blockIdx.x & 63;          // output row
  const int lane = tid & 31;
  const int wave = tid >> 5;
  const int wm   = wave & 3;                 // 4 M-waves * 16 rows
  const int wn   = wave >> 2;                // 2 N-waves * 128 cols

  // per-co tables: 2*||z||, cosh(2r)/||z||, sinh(2r)
  {
    int co = tid;
    float zn = sqrtf(fmaxf(zn2[co], 1e-30f));
    float e  = expf(2.0f * bias[co]);
    float ei = 1.0f / e;
    s_z2[co]   = 2.0f * zn;
    s_chzn[co] = 0.5f * (e + ei) / zn;
    s_sh[co]   = 0.5f * (e - ei);
  }
  // per-pixel tables: patch norm -> t = tanh(n)/n, lam = 2/(1-tanh^2)
  if (tid < 64) {
    int w = tid;
    float n2 = 0.f;
    for (int i = 0; i < 3; ++i) {
      int hh = h + i - 1;
      if (hh < 0 || hh > 63) continue;
      for (int j = 0; j < 3; ++j) {
        int ww = w + j - 1;
        if (ww < 0 || ww > 63) continue;
        n2 += s[(bimg << 12) + hh * 64 + ww];
      }
    }
    float n  = fmaxf(sqrtf(n2), 1e-15f);
    float th = tanhf(n);
    s_t[w]   = th / n;
    s_lam[w] = 2.0f / (1.0f - th * th);
  }
  // stage halo: s_v[r][wi][c] channel-contiguous, block-permuted + 32B-group
  // XOR swizzle by wi&15 (div-free incremental index update)
  {
    const int vbase = bimg << 20;
    int wi = tid % 66;
    int rc = tid / 66;
#pragma unroll 1
    for (int it = 0; it < 198; ++it) {       // 3*66*256 / 256 threads
      int r  = rc >> 8;
      int c  = rc & 255;
      int hh = h + r - 1;
      int ws = wi - 1;
      unsigned short bf = 0;
      if (hh >= 0 && hh < 64 && ws >= 0 && ws < 64)
        bf = vbf[vbase + (c << 12) + hh * 64 + ws];
      int cp  = (c & ~31) | (permblk((c >> 3) & 3) << 3) | (c & 7);
      int pos = (((cp >> 4) ^ (wi & 15)) << 4) | (cp & 15);
      s_v[(r * 66 + wi) * 256 + pos] = bf;
      wi += 58; rc += 3;                     // advance by 256 (= 3*66 + 58)
      if (wi >= 66) { wi -= 66; rc += 1; }
    }
  }

  v8f acc[8];
#pragma unroll
  for (int nt = 0; nt < 8; ++nt)
#pragma unroll
    for (int g = 0; g < 8; ++g) acc[nt][g] = 0.f;

  const int mrow  = wm * 16 + (lane & 15);   // pixel (w) owned by this lane
  const int klo16 = (lane < 16) ? 0 : 16;    // contiguous fragment start

  // software-pipelined B staging: regs hold next chunk while LDS serves current
  const int co0 = tid >> 2;                  // this thread's B rows: co0 + it*64
  const int kg  = tid & 3;                   // 8-half block within 32-K chunk
  const int pb  = permblk(kg);               // permuted store block
  v8us breg[4];
#pragma unroll
  for (int it = 0; it < 4; ++it)
    breg[it] = *(const v8us*)(Wt + (it * 64 + co0) * KTOT + kg * 8);

#pragma unroll 1
  for (int step = 0; step < 72; ++step) {    // K = 72 * 32
    __syncthreads();                         // prev compute done reading s_b
#pragma unroll
    for (int it = 0; it < 4; ++it)
      *(v8us*)(s_b + (it * 64 + co0) * BSTR + pb * 8) = breg[it];
    __syncthreads();                         // s_b ready
    if (step + 1 < 72) {
      int k0 = (step + 1) * 32;
#pragma unroll
      for (int it = 0; it < 4; ++it)
        breg[it] = *(const v8us*)(Wt + (it * 64 + co0) * KTOT + k0 + kg * 8);
    }

    // A fragment from resident halo: single contiguous 32B load
    int tap = step >> 3;
    int c0  = (step & 7) << 5;
    int di  = tap / 3;
    int dj  = tap - di * 3;
    int wi  = mrow + dj;
    int g16 = (c0 + klo16) >> 4;
    int aoff = ((di * 66 + wi) << 8) + ((g16 ^ (wi & 15)) << 4);
    v16bf a = __builtin_bit_cast(v16bf, *(const v16us*)(s_v + aoff));

#pragma unroll
    for (int nt = 0; nt < 8; ++nt) {
      int n = wn * 128 + nt * 16 + (lane & 15);
      v16bf bb = __builtin_bit_cast(v16bf, *(const v16us*)(s_b + n * BSTR + klo16));
      acc[nt] = __builtin_amdgcn_wmma_f32_16x16x32_bf16(
          false, a, false, bb, (short)0, acc[nt], false, false);
    }
  }

  // ---- epilogue ---------------------------------------------------------------
  // 1) dump raw accumulators to LDS (reuse halo buffer, padded 64x260 fp32)
  __syncthreads();
  float* wbuf = (float*)s_v;
  const int mbase = wm * 16 + ((lane >= 16) ? 8 : 0);
#pragma unroll
  for (int nt = 0; nt < 8; ++nt) {
    int n = wn * 128 + nt * 16 + (lane & 15);
#pragma unroll
    for (int g = 0; g < 8; ++g)
      wbuf[(mbase + g) * 260 + n] = acc[nt][g];
  }
  __syncthreads();

  // 2) rolled nonlinearity pass: one pixel row x 64 outputs per thread
  //    w = sinh(2|z| * asinh(inner)) = 0.5*(e^a - e^-a), a = 2|z|*log(q)
  int m = tid & 63, q = tid >> 6;
  float t   = s_t[m];
  float lam = s_lam[m];
  float lm1 = lam - 1.0f;
  float ps  = 0.f;
#pragma unroll 1
  for (int k = 0; k < 64; ++k) {
    int n = q * 64 + k;
    float xz    = t * wbuf[m * 260 + n];
    float inner = lam * xz * s_chzn[n] - lm1 * s_sh[n];
    float qq    = inner + sqrtf(inner * inner + 1.0f);
    float a     = s_z2[n] * logf(qq);
    float e     = expf(a);
    float wv    = 0.5f * (e - 1.0f / e);
    wbuf[m * 260 + n] = wv;
    ps += wv * wv;
  }
  s_ps[q * 64 + m] = ps;
  __syncthreads();
  if (tid < 64) {
    float tot = s_ps[tid] + s_ps[64 + tid] + s_ps[128 + tid] + s_ps[192 + tid];
    s_dinv[tid] = 1.0f / (1.0f + sqrtf(1.0f + tot));
  }
  __syncthreads();

  // 3) scaled store, coalesced along w
  float di = s_dinv[m];
  int obase = (bimg * COUT) * HW + h * 64 + m;
#pragma unroll 1
  for (int k = 0; k < 64; ++k) {
    int co = q * 64 + k;
    out[obase + co * HW] = wbuf[m * 260 + co] * di;
  }
}

// ------------------------------- host launcher --------------------------------
extern "C" void kernel_launch(void* const* d_in, const int* in_sizes, int n_in,
                              void* d_out, int out_size, void* d_ws, size_t ws_size,
                              hipStream_t stream) {
  const float* x    = (const float*)d_in[0];   // [16,256,64,64]
  const float* z    = (const float*)d_in[1];   // [2304,256]
  const float* bias = (const float*)d_in[2];   // [256]
  float* out = (float*)d_out;

  char* ws = (char*)d_ws;
  unsigned short* vbf  = (unsigned short*)(ws);              // 33,554,432 B
  float*          sbuf = (float*)(ws + 33554432);            //    262,144 B
  unsigned short* Wt   = (unsigned short*)(ws + 33816576);   //  1,179,648 B
  float*          part = (float*)(ws + 34996224);            //     73,728 B
  float*          zn2  = (float*)(ws + 35069952);            //      1,024 B

  // BETA(n/2,1/2)/BETA(n_i/2,1/2) computed on host (deterministic)
  double bni = std::exp(std::lgamma(128.0)  + std::lgamma(0.5) - std::lgamma(128.5));
  double bn  = std::exp(std::lgamma(1152.0) + std::lgamma(0.5) - std::lgamma(1152.5));
  float ratio = (float)(bn / bni);

  prep_pixels <<<256,  256, 0, stream>>>(x, vbf, sbuf, ratio);
  prep_weights<<<2304, 256, 0, stream>>>(z, Wt);
  znorm_part  <<<72,   256, 0, stream>>>(z, part);
  znorm_fin   <<<1,    256, 0, stream>>>(part, zn2);
  hconv_wmma  <<<1024, 256, 0, stream>>>(vbf, sbuf, Wt, zn2, bias, out);
}
